// MultiHeadAttention_33784212750781
// MI455X (gfx1250) — compile-verified, matrix operand order checked by reading, not run on hardware
//
#include <hip/hip_runtime.h>
#include <hip/hip_bf16.h>

// ---- problem constants ----
#define BATCH   2
#define SEQ     2048
#define DMODEL  1024
#define NHEAD   16
#define HDIM    64
#define MROWS   (BATCH * SEQ)          // 4096
#define SCALE_F 0.125f                 // 64^-0.5 (folded into Q at RoPE time)

typedef __bf16 bf16_t;
typedef __attribute__((ext_vector_type(16))) __bf16 bf16x16;
typedef __attribute__((ext_vector_type(8)))  __bf16 bf16x8;
typedef __attribute__((ext_vector_type(8)))  float  f32x8;
typedef unsigned int u32;
typedef __attribute__((ext_vector_type(4))) u32 u32x4;
typedef __attribute__((ext_vector_type(8))) u32 u32x8;

// ---------------------------------------------------------------------------
// Fragment loader: A-operand layout for v_wmma_*_16x16x32_bf16.
// lane r=lane%16 -> matrix row; sel=lane/16 picks K chunks {sel*8.., 16+sel*8..}
// Works for the B operand too when B is supplied as N x K row-major.
// ---------------------------------------------------------------------------
__device__ __forceinline__ bf16x16 load_frag(const bf16_t* base, int ld) {
    int lane = threadIdx.x & 31;
    int r    = lane & 15;
    int sel  = lane >> 4;
    const bf16_t* p = base + (size_t)r * ld + sel * 8;
    union { bf16x16 v; bf16x8 h[2]; } u;
    u.h[0] = *reinterpret_cast<const bf16x8*>(p);
    u.h[1] = *reinterpret_cast<const bf16x8*>(p + 16);
    return u.v;
}

__device__ __forceinline__ f32x8 wmma_bf16(bf16x16 a, bf16x16 b, f32x8 c) {
    return __builtin_amdgcn_wmma_f32_16x16x32_bf16(false, a, false, b, (short)0, c, false, false);
}

// ---------------------------------------------------------------------------
// Butterfly max within 16-lane rows using v_permlane16_b32 (pure VALU; avoids
// the ds_bpermute lowering of __shfl_xor and its DScnt stalls).
// ---------------------------------------------------------------------------
__device__ __forceinline__ float permlane16_max(float v, u32 sel_lo, u32 sel_hi) {
    union { float f; u32 u; } a, b;
    a.f = v;
    b.u = __builtin_amdgcn_permlane16(a.u, a.u, sel_lo, sel_hi, false, false);
    return fmaxf(v, b.f);
}

__device__ __forceinline__ float rowmax16(float v) {
    v = permlane16_max(v, 0x67452301u, 0xEFCDAB89u);   // xor 1
    v = permlane16_max(v, 0x54761032u, 0xDCFE98BAu);   // xor 2
    v = permlane16_max(v, 0x32107654u, 0xBA98FEDCu);   // xor 4
    v = permlane16_max(v, 0xFEDCBA98u, 0x76543210u);   // xor 8
    return v;
}

// ---------------------------------------------------------------------------
// TDM: issue one 2D tensor_load_to_lds (bf16 elements).
// Descriptor per CDNA5 ISA 8.3/8.4 (D# group0 128b, group1 256b), 2D tensor.
// ---------------------------------------------------------------------------
__device__ __forceinline__ void tdm_load_2d(const bf16_t* gptr, bf16_t* lptr,
                                            u32 rows, u32 tile_k, u32 ldk) {
    unsigned long long ga = (unsigned long long)(uintptr_t)gptr;
    u32 lds = (u32)(uintptr_t)lptr;               // low 32 bits of flat = LDS byte addr
    u32x4 g0;
    g0[0] = 1u;                                   // count=1, user descriptor
    g0[1] = lds;                                  // lds_addr
    g0[2] = (u32)ga;                              // global_addr[31:0]
    g0[3] = ((u32)(ga >> 32) & 0x01FFFFFFu) | 0x80000000u;  // addr[56:32] | type=2
    u32x8 g1;
    g1[0] = 0x00010000u;                          // data_size=1 (2 bytes)
    g1[1] = (ldk & 0xFFFFu) << 16;                // tensor_dim0[15:0]
    g1[2] = ((ldk >> 16) & 0xFFFFu) | ((rows & 0xFFFFu) << 16);   // td0 hi | td1 lo
    g1[3] = ((rows >> 16) & 0xFFFFu) | ((tile_k & 0xFFFFu) << 16);// td1 hi | tile_dim0
    g1[4] = rows & 0xFFFFu;                       // tile_dim1 (tile_dim2=0)
    g1[5] = ldk;                                  // tensor_dim0_stride[31:0]
    g1[6] = 0u;
    g1[7] = 0u;
    asm volatile("tensor_load_to_lds %0, %1" :: "s"(g0), "s"(g1) : "memory");
}

// ---------------------------------------------------------------------------
// fp32 -> bf16 elementwise convert
// ---------------------------------------------------------------------------
__global__ void k_f2bf(const float* __restrict__ in, bf16_t* __restrict__ out, int n) {
    int i = blockIdx.x * blockDim.x + threadIdx.x;
    if (i < n) out[i] = (bf16_t)in[i];
}

// ---------------------------------------------------------------------------
// GEMM:  C[M,N] = A[M,K] * B[N,K]^T   (bf16 in, f32 accum, OutT out)
// 256 threads = 8 waves; wave tile 64x64 (16 WMMA / k-step), WG tile 128x256.
// A/B panels staged into double-buffered LDS by the Tensor Data Mover.
// ---------------------------------------------------------------------------
template <typename OutT>
__global__ __launch_bounds__(256) void k_gemm_nt(const bf16_t* __restrict__ A,
                                                 const bf16_t* __restrict__ B,
                                                 OutT* __restrict__ C,
                                                 int M, int N, int K) {
    __shared__ __align__(16) bf16_t sa[2][128 * 32];   // A panel: 128 rows x 32 k
    __shared__ __align__(16) bf16_t sb[2][256 * 32];   // B panel: 256 rows x 32 k

    int wave = threadIdx.x >> 5;
    int lane = threadIdx.x & 31;
    int wr = wave >> 2, wc = wave & 3;                 // 2 x 4 waves
    int rowWG = blockIdx.x * 128;
    int colWG = blockIdx.y * 256;
    int row0 = rowWG + wr * 64;
    int col0 = colWG + wc * 64;

    f32x8 acc[4][4] = {};

    const int nIter = K / 32;
    if (wave == 0) {
        tdm_load_2d(A + (size_t)rowWG * K + 0,  sa[0], 128, 32, K);
        tdm_load_2d(B + (size_t)colWG * K + 0,  sb[0], 256, 32, K);
        tdm_load_2d(A + (size_t)rowWG * K + 32, sa[1], 128, 32, K);
        tdm_load_2d(B + (size_t)colWG * K + 32, sb[1], 256, 32, K);
    }

    for (int it = 0; it < nIter; ++it) {
        int cur = it & 1;
        if (wave == 0) {
            if (it + 1 < nIter) __builtin_amdgcn_s_wait_tensorcnt(2);  // cur pair done
            else                __builtin_amdgcn_s_wait_tensorcnt(0);  // last pair done
        }
        __syncthreads();

        bf16x16 af[4], bfm[4];
#pragma unroll
        for (int i = 0; i < 4; ++i)
            af[i] = load_frag(&sa[cur][(wr * 64 + i * 16) * 32], 32);
#pragma unroll
        for (int j = 0; j < 4; ++j)
            bfm[j] = load_frag(&sb[cur][(wc * 64 + j * 16) * 32], 32);
#pragma unroll
        for (int i = 0; i < 4; ++i)
#pragma unroll
            for (int j = 0; j < 4; ++j)
                acc[i][j] = wmma_bf16(af[i], bfm[j], acc[i][j]);

        __syncthreads();   // everyone done reading buf[cur] before TDM overwrites it
        if (wave == 0 && it + 2 < nIter) {
            int k2 = (it + 2) * 32;
            tdm_load_2d(A + (size_t)rowWG * K + k2, sa[cur], 128, 32, K);
            tdm_load_2d(B + (size_t)colWG * K + k2, sb[cur], 256, 32, K);
        }
    }

    // C/D layout: VGPR i, lane -> row = i + 8*(lane/16), col = lane%16
    int n = lane & 15, half = lane >> 4;
#pragma unroll
    for (int ti = 0; ti < 4; ++ti)
#pragma unroll
        for (int tj = 0; tj < 4; ++tj)
#pragma unroll
            for (int i = 0; i < 8; ++i) {
                int r = row0 + ti * 16 + i + 8 * half;
                int c = col0 + tj * 16 + n;
                C[(size_t)r * N + c] = (OutT)acc[ti][tj][i];
            }
}

// ---------------------------------------------------------------------------
// RoPE + transpose [B,S,H,Dh] -> [B,H,S,Dh] for Q and K (scale folded into Q).
// ---------------------------------------------------------------------------
__global__ void k_rope(const bf16_t* __restrict__ qlin, const bf16_t* __restrict__ klin,
                       bf16_t* __restrict__ qhd, bf16_t* __restrict__ khd) {
    int idx = blockIdx.x * blockDim.x + threadIdx.x;   // over B*S*H*32
    if (idx >= BATCH * SEQ * NHEAD * 32) return;
    int d = idx & 31;
    int t = idx >> 5;
    int h = t % NHEAD; t /= NHEAD;
    int s = t % SEQ;
    int b = t / SEQ;
    size_t iin  = (((size_t)b * SEQ + s) * NHEAD + h) * HDIM;
    size_t iout = (((size_t)b * NHEAD + h) * SEQ + s) * HDIM;
    float ang = (float)s * __powf(10000.0f, -(float)d * (1.0f / 32.0f));
    float sn, cs;
    __sincosf(ang, &sn, &cs);

    float q1 = (float)qlin[iin + d], q2 = (float)qlin[iin + d + 32];
    qhd[iout + d]      = (bf16_t)((q1 * cs - q2 * sn) * SCALE_F);
    qhd[iout + d + 32] = (bf16_t)((q2 * cs + q1 * sn) * SCALE_F);

    float k1 = (float)klin[iin + d], k2 = (float)klin[iin + d + 32];
    khd[iout + d]      = (bf16_t)(k1 * cs - k2 * sn);
    khd[iout + d + 32] = (bf16_t)(k2 * cs + k1 * sn);
}

// ---------------------------------------------------------------------------
// V transpose: [B,S,H,Dh] -> [B,H,Dh,S]  (so PV's B-operand loads like NxK)
// ---------------------------------------------------------------------------
__global__ void k_vtrans(const bf16_t* __restrict__ vlin, bf16_t* __restrict__ vt) {
    int idx = blockIdx.x * blockDim.x + threadIdx.x;   // over B*S*H*64
    if (idx >= BATCH * SEQ * NHEAD * HDIM) return;
    int d = idx & 63;
    int t = idx >> 6;
    int h = t % NHEAD; t /= NHEAD;
    int s = t % SEQ;
    int b = t / SEQ;
    vt[(((size_t)b * NHEAD + h) * HDIM + d) * SEQ + s] =
        vlin[(((size_t)b * SEQ + s) * NHEAD + h) * HDIM + d];
}

// ---------------------------------------------------------------------------
// Causal flash attention. block = 8 waves; each wave owns one 16-row Q tile,
// streams key tiles (32 keys / iter = two 16x16 score tiles), online softmax.
//  - unmasked main loop + one masked diagonal iteration
//  - row max via v_permlane16_b32 butterflies (VALU, no LDS traffic)
//  - row sum via an extra WMMA against a ones matrix (l kept in C-layout)
//  - P bounced through per-wave LDS to reshape C-layout -> A-fragment
// ---------------------------------------------------------------------------
__global__ __launch_bounds__(256) void k_flash(const bf16_t* __restrict__ qhd,
                                               const bf16_t* __restrict__ khd,
                                               const bf16_t* __restrict__ vt,
                                               bf16_t* __restrict__ obf) {
    __shared__ __align__(16) bf16_t pbuf[8][16 * 32];
    int wave = threadIdx.x >> 5;
    int lane = threadIdx.x & 31;
    int n = lane & 15, half = lane >> 4;

    int b = blockIdx.z, h = blockIdx.y;
    int q0 = (blockIdx.x * 8 + wave) * 16;

    const bf16_t* Q  = qhd + ((size_t)b * NHEAD + h) * SEQ * HDIM;
    const bf16_t* Kp = khd + ((size_t)b * NHEAD + h) * SEQ * HDIM;
    const bf16_t* Vt = vt  + ((size_t)b * NHEAD + h) * HDIM * SEQ;
    bf16_t* pw = &pbuf[wave][0];

    bf16x16 aq0 = load_frag(Q + (size_t)q0 * HDIM,      HDIM);
    bf16x16 aq1 = load_frag(Q + (size_t)q0 * HDIM + 32, HDIM);

    bf16x16 ones;
#pragma unroll
    for (int i = 0; i < 16; ++i) ones[i] = (bf16_t)1.0f;

    f32x8 o[4] = {};
    f32x8 lvec = {};                     // per-row softmax denominators (C-layout)
    float m[8];
#pragma unroll
    for (int i = 0; i < 8; ++i) m[i] = -__builtin_inff();

    const int nkb     = (q0 + 16 + 31) / 32;   // number of 32-key tiles
    const int kb_diag = (nkb - 1) * 32;        // only tile needing the causal mask

    // ---- hot loop: fully-valid key tiles, no masking ----
    for (int kb = 0; kb < kb_diag; kb += 32) {
        f32x8 s0 = {}, s1 = {};
        s0 = wmma_bf16(aq0, load_frag(Kp + (size_t)kb * HDIM,             HDIM), s0);
        s0 = wmma_bf16(aq1, load_frag(Kp + (size_t)kb * HDIM + 32,        HDIM), s0);
        s1 = wmma_bf16(aq0, load_frag(Kp + (size_t)(kb + 16) * HDIM,      HDIM), s1);
        s1 = wmma_bf16(aq1, load_frag(Kp + (size_t)(kb + 16) * HDIM + 32, HDIM), s1);

        float alpha[8];
#pragma unroll
        for (int i = 0; i < 8; ++i) {
            float mx = rowmax16(fmaxf(s0[i], s1[i]));
            float mnew = fmaxf(m[i], mx);
            alpha[i] = __expf(m[i] - mnew);
            float p0 = __expf(s0[i] - mnew);
            float p1 = __expf(s1[i] - mnew);
            pw[(i + 8 * half) * 32 + n]      = (bf16_t)p0;
            pw[(i + 8 * half) * 32 + 16 + n] = (bf16_t)p1;
            m[i] = mnew;
        }
#pragma unroll
        for (int j = 0; j < 4; ++j)
#pragma unroll
            for (int i = 0; i < 8; ++i) o[j][i] *= alpha[i];
#pragma unroll
        for (int i = 0; i < 8; ++i) lvec[i] *= alpha[i];

        asm volatile("s_wait_dscnt 0" ::: "memory");   // P stores before reload
        bf16x16 pa = load_frag(pw, 32);
        lvec = wmma_bf16(pa, ones, lvec);              // l += rowsum(P)
#pragma unroll
        for (int j = 0; j < 4; ++j) {
            bf16x16 bv = load_frag(Vt + (size_t)(j * 16) * SEQ + kb, SEQ);
            o[j] = wmma_bf16(pa, bv, o[j]);
        }
    }

    // ---- diagonal tile: apply causal mask ----
    {
        const int kb = kb_diag;
        f32x8 s0 = {}, s1 = {};
        s0 = wmma_bf16(aq0, load_frag(Kp + (size_t)kb * HDIM,             HDIM), s0);
        s0 = wmma_bf16(aq1, load_frag(Kp + (size_t)kb * HDIM + 32,        HDIM), s0);
        s1 = wmma_bf16(aq0, load_frag(Kp + (size_t)(kb + 16) * HDIM,      HDIM), s1);
        s1 = wmma_bf16(aq1, load_frag(Kp + (size_t)(kb + 16) * HDIM + 32, HDIM), s1);

        float alpha[8];
#pragma unroll
        for (int i = 0; i < 8; ++i) {
            int qrow = q0 + i + 8 * half;
            float v0 = s0[i];
            float v1 = s1[i];
            if (kb + n > qrow)      v0 = -__builtin_inff();
            if (kb + 16 + n > qrow) v1 = -__builtin_inff();
            float mx = rowmax16(fmaxf(v0, v1));
            float mnew = fmaxf(m[i], mx);
            alpha[i] = __expf(m[i] - mnew);
            float p0 = __expf(v0 - mnew);
            float p1 = __expf(v1 - mnew);
            pw[(i + 8 * half) * 32 + n]      = (bf16_t)p0;
            pw[(i + 8 * half) * 32 + 16 + n] = (bf16_t)p1;
            m[i] = mnew;
        }
#pragma unroll
        for (int j = 0; j < 4; ++j)
#pragma unroll
            for (int i = 0; i < 8; ++i) o[j][i] *= alpha[i];
#pragma unroll
        for (int i = 0; i < 8; ++i) lvec[i] *= alpha[i];

        asm volatile("s_wait_dscnt 0" ::: "memory");
        bf16x16 pa = load_frag(pw, 32);
        lvec = wmma_bf16(pa, ones, lvec);
#pragma unroll
        for (int j = 0; j < 4; ++j) {
            bf16x16 bv = load_frag(Vt + (size_t)(j * 16) * SEQ + kb, SEQ);
            o[j] = wmma_bf16(pa, bv, o[j]);
        }
    }

    float inv[8];
#pragma unroll
    for (int i = 0; i < 8; ++i) inv[i] = 1.0f / lvec[i];
#pragma unroll
    for (int j = 0; j < 4; ++j)
#pragma unroll
        for (int i = 0; i < 8; ++i) {
            int qrow = q0 + i + 8 * half;
            int d = j * 16 + n;
            obf[((size_t)b * SEQ + qrow) * DMODEL + h * HDIM + d] = (bf16_t)(o[j][i] * inv[i]);
        }
}

// ---------------------------------------------------------------------------
extern "C" void kernel_launch(void* const* d_in, const int* in_sizes, int n_in,
                              void* d_out, int out_size, void* d_ws, size_t ws_size,
                              hipStream_t stream) {
    const float* x  = (const float*)d_in[0];
    const float* wq = (const float*)d_in[1];
    const float* wk = (const float*)d_in[2];
    const float* wv = (const float*)d_in[3];
    const float* wo = (const float*)d_in[4];
    // d_in[5] = attention_mask (causal, handled analytically)

    char* ws = (char*)d_ws;
    const size_t SZ_X  = (size_t)MROWS * DMODEL * 2;       // 8 MB bf16
    const size_t SZ_W  = (size_t)DMODEL * DMODEL * 2;      // 2 MB bf16
    bf16_t* xb  = (bf16_t*)(ws);
    bf16_t* wqb = (bf16_t*)(ws + SZ_X);
    bf16_t* wkb = (bf16_t*)(ws + SZ_X + SZ_W);
    bf16_t* wvb = (bf16_t*)(ws + SZ_X + 2 * SZ_W);
    bf16_t* wob = (bf16_t*)(ws + SZ_X + 3 * SZ_W);
    bf16_t* ql  = (bf16_t*)(ws + SZ_X + 4 * SZ_W);
    bf16_t* kl  = (bf16_t*)(ws + 2 * SZ_X + 4 * SZ_W);
    bf16_t* vl  = (bf16_t*)(ws + 3 * SZ_X + 4 * SZ_W);
    bf16_t* qh  = (bf16_t*)(ws + 4 * SZ_X + 4 * SZ_W);
    bf16_t* kh  = (bf16_t*)(ws + 5 * SZ_X + 4 * SZ_W);
    bf16_t* vt  = (bf16_t*)(ws + 6 * SZ_X + 4 * SZ_W);
    bf16_t* ob  = (bf16_t*)(ws + 7 * SZ_X + 4 * SZ_W);

    const int NX = MROWS * DMODEL;        // 4194304
    const int NW = DMODEL * DMODEL;       // 1048576

    k_f2bf<<<(NX + 255) / 256, 256, 0, stream>>>(x,  xb,  NX);
    k_f2bf<<<(NW + 255) / 256, 256, 0, stream>>>(wq, wqb, NW);
    k_f2bf<<<(NW + 255) / 256, 256, 0, stream>>>(wk, wkb, NW);
    k_f2bf<<<(NW + 255) / 256, 256, 0, stream>>>(wv, wvb, NW);
    k_f2bf<<<(NW + 255) / 256, 256, 0, stream>>>(wo, wob, NW);

    dim3 ggrid(MROWS / 128, DMODEL / 256);
    k_gemm_nt<bf16_t><<<ggrid, 256, 0, stream>>>(xb, wqb, ql, MROWS, DMODEL, DMODEL);
    k_gemm_nt<bf16_t><<<ggrid, 256, 0, stream>>>(xb, wkb, kl, MROWS, DMODEL, DMODEL);
    k_gemm_nt<bf16_t><<<ggrid, 256, 0, stream>>>(xb, wvb, vl, MROWS, DMODEL, DMODEL);

    int nrope = BATCH * SEQ * NHEAD * 32;
    k_rope<<<(nrope + 255) / 256, 256, 0, stream>>>(ql, kl, qh, kh);
    int nvt = BATCH * SEQ * NHEAD * HDIM;
    k_vtrans<<<(nvt + 255) / 256, 256, 0, stream>>>(vl, vt);

    dim3 fgrid(SEQ / 128, NHEAD, BATCH);
    k_flash<<<fgrid, 256, 0, stream>>>(qh, kh, vt, ob);

    k_gemm_nt<float><<<ggrid, 256, 0, stream>>>(ob, wob, (float*)d_out, MROWS, DMODEL, DMODEL);
}
